// CorticalColumnGroup_79731772883718
// MI455X (gfx1250) — matrix operand order repaired and checked by reading, not compiled, not run on hardware
//
#include <hip/hip_runtime.h>

typedef __attribute__((ext_vector_type(16))) __bf16 v16bf;
typedef __attribute__((ext_vector_type(8)))  __bf16 v8bf;
typedef __attribute__((ext_vector_type(8)))  float  v8f;
typedef __attribute__((ext_vector_type(4)))  unsigned v4u;
typedef __attribute__((ext_vector_type(8)))  int      v8i;
typedef __attribute__((ext_vector_type(4)))  int      v4i;

#define DEVI __device__ __forceinline__

#if defined(__HIP_DEVICE_COMPILE__) && __has_builtin(__builtin_amdgcn_tensor_load_to_lds)
#define HAVE_TDM 1
#else
#define HAVE_TDM 0
#endif

constexpr int G = 128, DC = 128, DM = 64, DFF = 512;
constexpr int NSEQ = 512;
constexpr int RSLOT = 16, MSLOT = 64;
constexpr int TOK = 32;                 // tokens per workgroup tile (2 WMMA M-tiles)

// ---- d_out offsets (flat f32 elements, reference return order) ----
constexpr size_t O_Z     = 33554432;    // z
constexpr size_t O_ZHAT  = 50331648;    // z_hat
constexpr size_t O_SURP  = 67108864;    // surprise
constexpr size_t O_KCAND = 67371008;    // k_cand
constexpr size_t O_VPOST = 84148224;    // v_post
constexpr size_t O_GATE  = 100925440;   // gate
constexpr size_t O_QNOV  = 101187584;   // q_nov
constexpr size_t O_VCAND = 117964800;   // v_cand
constexpr size_t O_WNOV  = 134742016;   // w_nov
constexpr size_t O_S4    = 135004160;   // s4

// ---- workspace offsets (bf16 elements): prepped fragment-layout weights ----
constexpr size_t W_PMUP = 0;
constexpr size_t W_PMDN = W_PMUP + (size_t)G*DC*DM;
constexpr size_t W_EMUP = W_PMDN + (size_t)G*DM*DC;
constexpr size_t W_EMDN = W_EMUP + (size_t)G*DC*DM;
constexpr size_t W_ENC  = W_EMDN + (size_t)G*DM*DC;
constexpr size_t W_GAIN = W_ENC  + (size_t)G*DC*DM;
constexpr size_t W_FFU  = W_GAIN + (size_t)G*DM*DC;
constexpr size_t W_FFD  = W_FFU  + (size_t)G*DC*DFF;
constexpr size_t W_PRED = W_FFD  + (size_t)G*DFF*DC;
constexpr size_t W_POST = W_PRED + (size_t)G*DM*DM;   // first 256 cols of post_w

DEVI __bf16 f2bf(float f) {
  unsigned u = __builtin_bit_cast(unsigned, f);
  unsigned r = u + 0x7FFFu + ((u >> 16) & 1u);        // RNE
  unsigned short h = (unsigned short)(r >> 16);
  return __builtin_bit_cast(__bf16, h);
}

#if HAVE_TDM
// Tensor Data Mover: 2D tile (tile_d1 rows x tile_d0 elems, 4B elems) from a
// strided global tensor into contiguous LDS at lds_off. D# built per ISA ch.8:
//   group0: [1:0]count=1 | [63:32]lds_addr | [120:64]global_addr | [127:126]type=2
//   group1: [17:16]data_size=2(4B) | tensor_dim0[79:48] | tensor_dim1[111:80]
//           | tile_dim0[127:112] | tile_dim1[143:128] | tensor_dim0_stride[207:160]
//   groups 2/3 (and trailing group) zero -> 2D tensor, dims above tile_dim1 unused
DEVI void tdm_load_2d(const void* gptr, unsigned lds_off,
                      unsigned tile_d0, unsigned tile_d1,
                      unsigned long long stride0_elems) {
  unsigned long long ga = (unsigned long long)gptr;
  v4u g0;
  g0[0] = 1u;                                          // count=1, user descriptor
  g0[1] = lds_off;                                     // LDS byte address
  g0[2] = (unsigned)(ga & 0xFFFFFFFFu);                // global_addr[31:0]
  g0[3] = (unsigned)((ga >> 32) & 0x01FFFFFFu) | (2u << 30); // addr[56:32], type=2
  v8i g1;
  g1[0] = (int)(2u << 16);                             // wg_mask=0, data_size=4B
  g1[1] = (int)((tile_d0 & 0xFFFFu) << 16);            // tensor_dim0 = tile_d0
  g1[2] = (int)(((tile_d0 >> 16) & 0xFFFFu) | ((tile_d1 & 0xFFFFu) << 16)); // tensor_dim1 = tile_d1
  g1[3] = (int)(((tile_d1 >> 16) & 0xFFFFu) | ((tile_d0 & 0xFFFFu) << 16)); // tile_dim0
  g1[4] = (int)(tile_d1 & 0xFFFFu);                    // tile_dim1; tile_dim2=0
  g1[5] = (int)(unsigned)(stride0_elems & 0xFFFFFFFFu);        // dim0 stride lo
  g1[6] = (int)(unsigned)((stride0_elems >> 32) & 0xFFFFu);    // dim0 stride hi
  g1[7] = 0;                                           // dim1 stride unused (2D)
  v4i gz4 = {0,0,0,0};
  v8i gz8 = {0,0,0,0,0,0,0,0};
  __builtin_amdgcn_tensor_load_to_lds(g0, g1, gz4, gz4, gz8, 0);
}
#endif

// Convert (G,K,srcN) f32 weights to bf16 B-fragment layout:
// fragment(nt,kt): 32 lanes x 16 contiguous bf16; lane's element j =
//   W[kt*32 + (lane>>4)*16 + j][nt*16 + (lane&15)]  (contiguous k order, same as the A loads)
__global__ void prep_w(const float* __restrict__ src, __bf16* __restrict__ dst,
                       int K, int NT, int srcN) {
  int g = blockIdx.x, t = blockIdx.y;
  int KT = K >> 5;
  int kt = t % KT, nt = t / KT;
  int lane = threadIdx.x;
  int half = lane >> 4, col = lane & 15;
  const float* s = src + (size_t)g * K * srcN;
  __bf16* d = dst + (size_t)g * K * (NT * 16) + ((size_t)(nt * KT + kt) * 32 + lane) * 16;
  int k0 = kt * 32 + half * 16;
  int n  = nt * 16 + col;
  #pragma unroll
  for (int j = 0; j < 16; ++j) d[j] = f2bf(s[(size_t)(k0 + j) * srcN + n]);
}

// One 16x16 output tile, K-loop over 32-wide steps, f32 accumulation.
// A: LDS, row-major [M][K] bf16 (k = 16*half + s: one contiguous 32B run/lane, matching prep_w's B order).
template<int K>
DEVI v8f wmma_tile(const __bf16* __restrict__ Bf, const __bf16* __restrict__ A,
                   int mt, int nt, int lane) {
  constexpr int KT = K >> 5;
  const int half = lane >> 4, row = lane & 15;
  const __bf16* arow = A + (mt * 16 + row) * K + half * 16;
  const __bf16* bfr  = Bf + ((size_t)(nt * KT) * 32 + lane) * 16;
  v8f acc = {0.f,0.f,0.f,0.f,0.f,0.f,0.f,0.f};
  #pragma unroll
  for (int kt = 0; kt < KT; ++kt) {
    union { v16bf v; v8bf h[2]; } a;
    a.h[0] = *(const v8bf*)(arow + kt * 32);
    a.h[1] = *(const v8bf*)(arow + kt * 32 + 8);
    v16bf b = *(const v16bf*)(bfr + (size_t)kt * 512);
    acc = __builtin_amdgcn_wmma_f32_16x16x32_bf16(false, a.v, false, b,
                                                  (short)0, acc, false, false);
  }
  return acc;
}

// Distribute 2*NT output tiles over 8 waves; `post` consumes each tile's acc.
template<int K, int NT, typename F>
DEVI void gemm_stage(const __bf16* __restrict__ Bf, const __bf16* __restrict__ A,
                     int wave, int lane, F post) {
  for (int t = wave; t < 2 * NT; t += 8) {
    int mt = t / NT, nt = t % NT;
    v8f acc = wmma_tile<K>(Bf, A, mt, nt, lane);
    post(mt, nt, lane >> 4, lane & 15, acc);
  }
}

__global__ __launch_bounds__(256) void ccg_main(
    const float* __restrict__ x_col, const float* __restrict__ pm_state,
    const float* __restrict__ em_state, const float* __restrict__ z_hat_prev,
    const float* __restrict__ ffn_norm_w, const float* __restrict__ ffn_norm_b,
    const float* __restrict__ ffn_up_b, const float* __restrict__ ffn_down_b,
    const float* __restrict__ pm_up_b, const float* __restrict__ pm_down_b,
    const float* __restrict__ em_up_b, const float* __restrict__ em_down_b,
    const float* __restrict__ post_w, const float* __restrict__ post_b,
    const float* __restrict__ enc_b, const float* __restrict__ pred_b,
    const float* __restrict__ gain_b, const __bf16* __restrict__ wsb,
    float* __restrict__ out)
{
  __shared__ __align__(16) float  xf[TOK*DC];     // activation f32 (TDM dest)
  __shared__ __align__(16) float  zhb[TOK*DM];    // z_hat_prev tile (TDM dest)
  __shared__ __align__(16) __bf16 xb[TOK*DC];     // activation bf16 (WMMA A)
  __shared__ __align__(16) float  qf[TOK*DM];     // q_pm / q_em / delta f32 / k_pre
  __shared__ __align__(16) float  sbuf[TOK*MSLOT];// attention scores / k_cand_raw
  __shared__ __align__(16) __bf16 mrb[TOK*DM];    // mem_read out bf16, later z bf16
  __shared__ __align__(16) __bf16 db[TOK*DM];     // delta bf16
  __shared__ __align__(16) float  gainf[TOK*DC];
  __shared__ __align__(16) __bf16 hb[TOK*DC];     // gln(x)*gain bf16
  __shared__ __align__(16) __bf16 ub[TOK*DFF];    // gelu(ffn_up) bf16
  __shared__ float redA[TOK];
  __shared__ float redB[TOK];

  const int tid  = threadIdx.x;
  const int wave = tid >> 5, lane = tid & 31;
  const int g  = blockIdx.x;
  const int T0 = blockIdx.y * TOK;      // tile never crosses a batch boundary
  const int bs = T0 / NSEQ;
  const int bb = g >> 4;                // g = b*16 + c

  // ---- stage activation tiles into LDS ----
#if HAVE_TDM
  if (wave == 0) {
    // x_col tile: 32 rows x 128 f32, row stride G*DC elems -> contiguous 16KB LDS
    tdm_load_2d(x_col + ((size_t)T0 * G + g) * DC,
                (unsigned)(size_t)(xf), DC, TOK, (unsigned long long)G * DC);
    // z_hat_prev tile: 32 rows x 64 f32, row stride G*DM elems -> 8KB LDS
    tdm_load_2d(z_hat_prev + ((size_t)T0 * G + g) * DM,
                (unsigned)(size_t)(zhb), DM, TOK, (unsigned long long)G * DM);
  }
  // warm L2/L0 for the first WMMA weight streams while the TDM runs
  __builtin_prefetch(wsb + W_PMUP + (size_t)g*DC*DM + (size_t)tid*64, 0, 1);
  __builtin_prefetch(wsb + W_FFU  + (size_t)g*DC*DFF + (size_t)tid*64, 0, 1);
  if (wave == 0) __builtin_amdgcn_s_wait_tensorcnt(0);
  __syncthreads();
  for (int i = tid; i < TOK*DC; i += 256) xb[i] = f2bf(xf[i]);
  __syncthreads();
#else
  #pragma unroll
  for (int i = 0; i < 4; ++i) {
    int idx = tid + i * 256;
    int row = idx >> 5;
    int c4  = (idx & 31) << 2;
    const float4 v = *(const float4*)(x_col + ((size_t)(T0 + row) * G + g) * DC + c4);
    float* xr = xf + row * DC + c4;
    xr[0]=v.x; xr[1]=v.y; xr[2]=v.z; xr[3]=v.w;
    __bf16* br = xb + row * DC + c4;
    br[0]=f2bf(v.x); br[1]=f2bf(v.y); br[2]=f2bf(v.z); br[3]=f2bf(v.w);
  }
  for (int i = tid; i < TOK*DM; i += 256) {
    int m = i >> 6, d = i & 63;
    zhb[i] = z_hat_prev[((size_t)(T0 + m) * G + g) * DM + d];
  }
  __syncthreads();
#endif

  // ---- q_pm = x @ pm_up + b ----
  gemm_stage<DC, DM/16>(wsb + W_PMUP + (size_t)g*DC*DM, xb, wave, lane,
    [&](int mt, int nt, int half, int col, v8f acc) {
      int c = nt*16 + col; float bv = pm_up_b[g*DM + c];
      #pragma unroll
      for (int q = 0; q < 8; ++q) qf[(mt*16 + half*8 + q)*DM + c] = acc[q] + bv;
    });
  __syncthreads();

  // ---- pm attention (R=16 slots) ----
  {
    const float* st = pm_state + (size_t)(bs*8 + bb) * RSLOT * DM;
    for (int s = tid; s < TOK*RSLOT; s += 256) {
      int tok = s >> 4, r = s & 15;
      float a = 0.f;
      for (int d = 0; d < DM; ++d) a += qf[tok*DM + d] * st[r*DM + d];
      sbuf[tok*MSLOT + r] = a * 0.125f;
    }
    __syncthreads();
    if (tid < TOK) {
      float mx = -1e30f;
      for (int r = 0; r < RSLOT; ++r) mx = fmaxf(mx, sbuf[tid*MSLOT + r]);
      float sm = 0.f;
      for (int r = 0; r < RSLOT; ++r) { float e = expf(sbuf[tid*MSLOT + r] - mx); sbuf[tid*MSLOT + r] = e; sm += e; }
      float inv = 1.f / sm;
      for (int r = 0; r < RSLOT; ++r) sbuf[tid*MSLOT + r] *= inv;
    }
    __syncthreads();
    for (int o = tid; o < TOK*DM; o += 256) {
      int tok = o >> 6, d = o & 63;
      float a = 0.f;
      for (int r = 0; r < RSLOT; ++r) a += sbuf[tok*MSLOT + r] * st[r*DM + d];
      mrb[o] = f2bf(a);
    }
    __syncthreads();
  }

  // ---- x += mr @ pm_down + b ----
  gemm_stage<DM, DC/16>(wsb + W_PMDN + (size_t)g*DM*DC, mrb, wave, lane,
    [&](int mt, int nt, int half, int col, v8f acc) {
      int c = nt*16 + col; float bv = pm_down_b[g*DC + c];
      #pragma unroll
      for (int q = 0; q < 8; ++q) xf[(mt*16 + half*8 + q)*DC + c] += acc[q] + bv;
    });
  __syncthreads();
  for (int i = tid; i < TOK*DC; i += 256) xb[i] = f2bf(xf[i]);
  __syncthreads();

  // ---- q_em = x @ em_up + b ----
  gemm_stage<DC, DM/16>(wsb + W_EMUP + (size_t)g*DC*DM, xb, wave, lane,
    [&](int mt, int nt, int half, int col, v8f acc) {
      int c = nt*16 + col; float bv = em_up_b[g*DM + c];
      #pragma unroll
      for (int q = 0; q < 8; ++q) qf[(mt*16 + half*8 + q)*DM + c] = acc[q] + bv;
    });
  __syncthreads();

  // ---- em attention (M=64 slots) ----
  {
    const float* st = em_state + (size_t)(bs*8 + bb) * MSLOT * DM;
    for (int s = tid; s < TOK*MSLOT; s += 256) {
      int tok = s >> 6, r = s & 63;
      float a = 0.f;
      for (int d = 0; d < DM; ++d) a += qf[tok*DM + d] * st[r*DM + d];
      sbuf[tok*MSLOT + r] = a * 0.125f;
    }
    __syncthreads();
    if (tid < TOK) {
      float mx = -1e30f;
      for (int r = 0; r < MSLOT; ++r) mx = fmaxf(mx, sbuf[tid*MSLOT + r]);
      float sm = 0.f;
      for (int r = 0; r < MSLOT; ++r) { float e = expf(sbuf[tid*MSLOT + r] - mx); sbuf[tid*MSLOT + r] = e; sm += e; }
      float inv = 1.f / sm;
      for (int r = 0; r < MSLOT; ++r) sbuf[tid*MSLOT + r] *= inv;
    }
    __syncthreads();
    for (int o = tid; o < TOK*DM; o += 256) {
      int tok = o >> 6, d = o & 63;
      float a = 0.f;
      for (int r = 0; r < MSLOT; ++r) a += sbuf[tok*MSLOT + r] * st[r*DM + d];
      mrb[o] = f2bf(a);
    }
    __syncthreads();
  }

  // ---- x += mr @ em_down + b ----
  gemm_stage<DM, DC/16>(wsb + W_EMDN + (size_t)g*DM*DC, mrb, wave, lane,
    [&](int mt, int nt, int half, int col, v8f acc) {
      int c = nt*16 + col; float bv = em_down_b[g*DC + c];
      #pragma unroll
      for (int q = 0; q < 8; ++q) xf[(mt*16 + half*8 + q)*DC + c] += acc[q] + bv;
    });
  __syncthreads();
  for (int i = tid; i < TOK*DC; i += 256) xb[i] = f2bf(xf[i]);
  __syncthreads();

  // ---- z = x @ enc + b ; delta = z - z_hat_prev ----
  gemm_stage<DC, DM/16>(wsb + W_ENC + (size_t)g*DC*DM, xb, wave, lane,
    [&](int mt, int nt, int half, int col, v8f acc) {
      int c = nt*16 + col; float bv = enc_b[g*DM + c];
      #pragma unroll
      for (int q = 0; q < 8; ++q) {
        int m = mt*16 + half*8 + q; int T = T0 + m;
        float zv = acc[q] + bv;
        out[O_Z + ((size_t)T*G + g)*DM + c] = zv;
        mrb[m*DM + c] = f2bf(zv);                          // z bf16 (for pred)
        float dl = zv - zhb[m*DM + c];
        qf[m*DM + c] = dl;
        db[m*DM + c] = f2bf(dl);
      }
    });
  __syncthreads();

  // ---- surprise / gate / s4 ----
  if (tid < TOK) {
    float s2 = 0.f;
    for (int d = 0; d < DM; ++d) { float v = qf[tid*DM + d]; s2 += v*v; }
    float s = sqrtf(s2);
    int T = T0 + tid;
    size_t o = (size_t)T*G + g;
    out[O_SURP + o] = s;
    out[O_S4   + o] = s;
    out[O_GATE + o] = fminf(fmaxf(s, 0.f), 1.f);
  }

  // ---- gain = 1 + 0.1*tanh(delta @ gain_w + b) ----
  gemm_stage<DM, DC/16>(wsb + W_GAIN + (size_t)g*DM*DC, db, wave, lane,
    [&](int mt, int nt, int half, int col, v8f acc) {
      int c = nt*16 + col; float bv = gain_b[g*DC + c];
      #pragma unroll
      for (int q = 0; q < 8; ++q)
        gainf[(mt*16 + half*8 + q)*DC + c] = 1.0f + 0.1f*tanhf(acc[q] + bv);
    });
  __syncthreads();

  // ---- h0 = gln(x) * gain ----
  if (tid < TOK) {
    float mu = 0.f;
    for (int c = 0; c < DC; ++c) mu += xf[tid*DC + c];
    mu *= (1.f/DC);
    float var = 0.f;
    for (int c = 0; c < DC; ++c) { float d_ = xf[tid*DC + c] - mu; var += d_*d_; }
    var *= (1.f/DC);
    redA[tid] = mu;
    redB[tid] = rsqrtf(var + 1e-5f);
  }
  __syncthreads();
  for (int i = tid; i < TOK*DC; i += 256) {
    int m = i >> 7, c = i & 127;
    float hv = (xf[i] - redA[m]) * redB[m] * ffn_norm_w[g*DC + c] + ffn_norm_b[g*DC + c];
    hb[i] = f2bf(hv * gainf[i]);
  }
  __syncthreads();

  // ---- u = gelu(h0 @ ffn_up + b) ----
  gemm_stage<DC, DFF/16>(wsb + W_FFU + (size_t)g*DC*DFF, hb, wave, lane,
    [&](int mt, int nt, int half, int col, v8f acc) {
      int c = nt*16 + col; float bv = ffn_up_b[g*DFF + c];
      #pragma unroll
      for (int q = 0; q < 8; ++q) {
        float v = acc[q] + bv;
        ub[(mt*16 + half*8 + q)*DFF + c] = f2bf(0.5f*v*(1.f + erff(v*0.70710678118f)));
      }
    });
  __syncthreads();

  // ---- x_out = x + u @ ffn_down + b ----
  gemm_stage<DFF, DC/16>(wsb + W_FFD + (size_t)g*DFF*DC, ub, wave, lane,
    [&](int mt, int nt, int half, int col, v8f acc) {
      int c = nt*16 + col; float bv = ffn_down_b[g*DC + c];
      #pragma unroll
      for (int q = 0; q < 8; ++q) {
        int m = mt*16 + half*8 + q; int T = T0 + m;
        float xo = xf[m*DC + c] + acc[q] + bv;
        out[((size_t)T*G + g)*DC + c] = xo;
        xf[m*DC + c] = xo;
      }
    });
  __syncthreads();
  for (int i = tid; i < TOK*DC; i += 256) xb[i] = f2bf(xf[i]);
  __syncthreads();

  // ---- z_hat = z @ pred + b ----
  gemm_stage<DM, DM/16>(wsb + W_PRED + (size_t)g*DM*DM, mrb, wave, lane,
    [&](int mt, int nt, int half, int col, v8f acc) {
      int c = nt*16 + col; float bv = pred_b[g*DM + c];
      #pragma unroll
      for (int q = 0; q < 8; ++q) {
        int m = mt*16 + half*8 + q; int T = T0 + m;
        out[O_ZHAT + ((size_t)T*G + g)*DM + c] = acc[q] + bv;
      }
    });

  // ---- proj = x_out @ post_w (first 256 cols) + b ----
  gemm_stage<DC, 16>(wsb + W_POST + (size_t)g*DC*256, xb, wave, lane,
    [&](int mt, int nt, int half, int col, v8f acc) {
      int c = nt*16 + col; float bv = post_b[g*257 + c];
      int chunk = c >> 6, d = c & 63;
      #pragma unroll
      for (int q = 0; q < 8; ++q) {
        int m = mt*16 + half*8 + q; int T = T0 + m;
        float v = acc[q] + bv;
        size_t o = ((size_t)T*G + g)*DM + d;
        if      (chunk == 0) qf[m*DM + d] = v;              // k_pre
        else if (chunk == 1) out[O_VPOST + o] = v;          // v_post
        else if (chunk == 2) sbuf[m*MSLOT + d] = v;         // k_cand_raw
        else                 out[O_VCAND + o] = v;          // v_cand
      }
    });
  __syncthreads();

  // ---- unit-normalize k_pre -> k_cand, k_cand_raw -> q_nov ----
  if (tid < 2*TOK) {
    int m = tid & (TOK-1);
    const float* bp = (tid < TOK) ? (qf + m*DM) : (sbuf + m*MSLOT);
    float s2 = 0.f;
    for (int d = 0; d < DM; ++d) s2 += bp[d]*bp[d];
    float inv = 1.f / (sqrtf(s2) + 1e-6f);
    if (tid < TOK) redA[m] = inv; else redB[m] = inv;
  }
  __syncthreads();
  for (int i = tid; i < TOK*DM; i += 256) {
    int m = i >> 6, d = i & 63;
    int T = T0 + m;
    size_t o = ((size_t)T*G + g)*DM + d;
    out[O_KCAND + o] = qf[m*DM + d] * redA[m];
    out[O_QNOV  + o] = sbuf[m*MSLOT + d] * redB[m];
  }

  // ---- nov_raw (post_w col 256) -> w_nov ----
  if (tid < TOK) {
    float a = post_b[g*257 + 256];
    const float* pw = post_w + (size_t)g * DC * 257 + 256;
    for (int k = 0; k < DC; ++k) a += xf[tid*DC + k] * pw[(size_t)k*257];
    int T = T0 + tid;
    out[O_WNOV + (size_t)T*G + g] = 1.f / (1.f + expf(-a));
  }
}

extern "C" void kernel_launch(void* const* d_in, const int* in_sizes, int n_in,
                              void* d_out, int out_size, void* d_ws, size_t ws_size,
                              hipStream_t stream) {
  (void)in_sizes; (void)n_in; (void)out_size; (void)ws_size;
  __bf16* wsb = (__bf16*)d_ws;

  auto W = [&](int idx, size_t off, int K, int NT, int srcN) {
    int KT = K >> 5;
    prep_w<<<dim3(G, KT*NT), dim3(32), 0, stream>>>(
        (const float*)d_in[idx], wsb + off, K, NT, srcN);
  };
  W(10, W_PMUP, 128,  4,  64);   // pm_up_w   (G,128,64)
  W(12, W_PMDN,  64,  8, 128);   // pm_down_w (G,64,128)
  W(14, W_EMUP, 128,  4,  64);   // em_up_w
  W(16, W_EMDN,  64,  8, 128);   // em_down_w
  W(20, W_ENC,  128,  4,  64);   // enc_w
  W(24, W_GAIN,  64,  8, 128);   // gain_w
  W(6,  W_FFU,  128, 32, 512);   // ffn_up_w
  W(8,  W_FFD,  512,  8, 128);   // ffn_down_w
  W(22, W_PRED,  64,  4,  64);   // pred_w
  W(18, W_POST, 128, 16, 257);   // post_w (first 256 cols)

  ccg_main<<<dim3(G, (4*NSEQ)/TOK), dim3(256), 0, stream>>>(
      (const float*)d_in[0],  (const float*)d_in[1],
      (const float*)d_in[2],  (const float*)d_in[3],
      (const float*)d_in[4],  (const float*)d_in[5],
      (const float*)d_in[7],  (const float*)d_in[9],
      (const float*)d_in[11], (const float*)d_in[13],
      (const float*)d_in[15], (const float*)d_in[17],
      (const float*)d_in[18], (const float*)d_in[19],
      (const float*)d_in[21], (const float*)d_in[23],
      (const float*)d_in[25],
      wsb, (float*)d_out);
}
